// RACModel_2602750182040
// MI455X (gfx1250) — matrix operand order repaired and checked by reading, not compiled, not run on hardware
//
#include <hip/hip_runtime.h>

// ---- problem constants (match reference) ----
constexpr int B_    = 2048;
constexpr int C_    = 96;
constexpr int D_    = 384;
constexpr int N_    = 200000;
constexpr int NCLS_ = 10;
constexpr float LN_EPS_ = 1e-5f;

// ---- WMMA vector types (gfx1250, wave32) ----
typedef __attribute__((ext_vector_type(16))) __bf16        v16bf;
typedef __attribute__((ext_vector_type(8)))  float         v8f;
typedef __attribute__((ext_vector_type(8)))  unsigned int  v8u;

union Frag { v8u u; v16bf b; };

// fp32 -> bf16 round-to-nearest-even
__device__ __forceinline__ unsigned short f2bf(float f) {
    union { float f; unsigned int u; } c; c.f = f;
    unsigned int u = c.u;
    u += 0x7FFFu + ((u >> 16) & 1u);
    return (unsigned short)(u >> 16);
}

__device__ __forceinline__ float wave_sum(float v) {
#pragma unroll
    for (int off = 16; off > 0; off >>= 1) v += __shfl_xor(v, off, 32);
    return v;
}
__device__ __forceinline__ float wave_max(float v) {
#pragma unroll
    for (int off = 16; off > 0; off >>= 1) v = fmaxf(v, __shfl_xor(v, off, 32));
    return v;
}
__device__ __forceinline__ float silu(float x) {
    return x / (1.0f + __expf(-x));
}

// ======================================================================
// Pack a fp32 row-major [K=384][N=384] weight into bf16 B-fragment layout.
// Fragment (nt,kt): 32 lanes x 8 dwords contiguous.
//   lane: col = lane&15, kbase = (lane>>4)*16
//   dword j: (W[k0+kbase+2j][n], W[k0+kbase+2j+1][n]) packed lo|hi
// total dwords = 24 ntiles * 12 ktiles * 32 lanes * 8 = 73728
// ======================================================================
__global__ void pack_w_kernel(const float* __restrict__ W,
                              unsigned int* __restrict__ out) {
    int t = blockIdx.x * 256 + threadIdx.x;
    if (t >= 24 * 12 * 32 * 8) return;
    int j    = t & 7;
    int rest = t >> 3;
    int lane = rest & 31;
    int tile = rest >> 5;           // 0..287
    int kt   = tile % 12;
    int nt   = tile / 12;
    int k    = kt * 32 + ((lane >> 4) << 4) + 2 * j;
    int n    = nt * 16 + (lane & 15);
    unsigned int lo = f2bf(W[k * D_ + n]);
    unsigned int hi = f2bf(W[(k + 1) * D_ + n]);
    out[t] = lo | (hi << 16);
}

// ======================================================================
// |candidate_embeddings[n]|^2 for all n.  one wave per row.
// ======================================================================
__global__ void cand_norms_kernel(const float* __restrict__ cand,
                                  float* __restrict__ cn) {
    int wave = threadIdx.x >> 5, lane = threadIdx.x & 31;
    int row = blockIdx.x * 8 + wave;
    if (row >= N_) return;
    const float* p = cand + (size_t)row * D_;
    float s = 0.f;
    for (int d = lane; d < D_; d += 32) { float v = p[d]; s += v * v; }
    s = wave_sum(s);
    if (lane == 0) cn[row] = s;
}

// ======================================================================
// sim[b,c] = -|k|^2 + 2 k.ck - |ck|^2 ; softmax over c -> probs[b,c]
// one block per b, 8 waves.
// ======================================================================
__global__ void sim_softmax_kernel(const float* __restrict__ k,
                                   const int*   __restrict__ cidx,
                                   const float* __restrict__ cand,
                                   const float* __restrict__ cn,
                                   float* __restrict__ probs) {
    __shared__ float kl[D_];
    __shared__ float sim[C_];
    int b = blockIdx.x;
    int tid = threadIdx.x, wave = tid >> 5, lane = tid & 31;

    for (int d = tid; d < D_; d += 256) kl[d] = k[(size_t)b * D_ + d];
    __syncthreads();

    float kk = 0.f;
    for (int d = lane; d < D_; d += 32) kk += kl[d] * kl[d];
    kk = wave_sum(kk);      // all lanes hold |k|^2

    for (int c = wave; c < C_; c += 8) {
        int idx = cidx[b * C_ + c];
        const float* p = cand + (size_t)idx * D_;
        float dot = 0.f;
        for (int d = lane; d < D_; d += 32) dot += kl[d] * p[d];
        dot = wave_sum(dot);
        if (lane == 0) sim[c] = -kk + 2.0f * dot - cn[idx];
    }
    __syncthreads();

    if (wave == 0) {   // 96 = 3 * 32 values
        float a0 = sim[lane], a1 = sim[lane + 32], a2 = sim[lane + 64];
        float m = wave_max(fmaxf(a0, fmaxf(a1, a2)));
        float e0 = __expf(a0 - m), e1 = __expf(a1 - m), e2 = __expf(a2 - m);
        float s = wave_sum(e0 + e1 + e2);
        float inv = 1.0f / s;
        probs[b * C_ + lane]      = e0 * inv;
        probs[b * C_ + lane + 32] = e1 * inv;
        probs[b * C_ + lane + 64] = e2 * inv;
    }
}

// ======================================================================
// Heavy kernel: per 16-row stripe (same b, 16 consecutive c):
//   xin = k - ck ; LN ; h1 = silu(xln @ W1 + b1) ; h2 = h1 @ W2 + b2
//   val = label_emb[y] + xin + h2 ; ctx[b] += sum_c probs * val
// block = 256 threads = 8 waves, each wave owns 3 N-tiles of 16.
// ======================================================================
__global__ void values_kernel(const float* __restrict__ k,
                              const int*   __restrict__ cidx,
                              const float* __restrict__ cand,
                              const int*   __restrict__ cy,
                              const float* __restrict__ lemb,
                              const float* __restrict__ lng,
                              const float* __restrict__ lnb,
                              const float* __restrict__ kpb1,
                              const float* __restrict__ kpb2,
                              const unsigned int* __restrict__ w1p,
                              const unsigned int* __restrict__ w2p,
                              const float* __restrict__ probs,
                              float* __restrict__ ctx_out) {
    __shared__ __align__(16) float          xinf[16 * D_];   // 24 KB (kept for residual)
    __shared__ __align__(16) unsigned short xlns[16 * D_];   // 12 KB bf16 (A of GEMM1)
    __shared__ __align__(16) unsigned short h1s [16 * D_];   // 12 KB bf16 (A of GEMM2)
    __shared__ float row_prob[16];
    __shared__ int   row_y[16];
    __shared__ int   row_idx[16];

    int b  = blockIdx.x / 6;
    int c0 = (blockIdx.x % 6) * 16;
    int tid = threadIdx.x, wave = tid >> 5, lane = tid & 31;

    if (tid < 16) {
        int idx = cidx[b * C_ + c0 + tid];
        row_idx[tid]  = idx;
        row_y[tid]    = cy[idx];
        row_prob[tid] = probs[b * C_ + c0 + tid];
    }
    __syncthreads();

    // gather: xin = k[b] - ck   (coalesced over d)
    for (int i = tid; i < 16 * D_; i += 256) {
        int r = i / D_, d = i - r * D_;
        xinf[i] = k[(size_t)b * D_ + d] - cand[(size_t)row_idx[r] * D_ + d];
    }
    __syncthreads();

    // LayerNorm: wave w handles rows 2w, 2w+1
#pragma unroll
    for (int rr = 0; rr < 2; ++rr) {
        int r = wave * 2 + rr;
        float s = 0.f, s2 = 0.f;
        for (int d = lane; d < D_; d += 32) {
            float v = xinf[r * D_ + d]; s += v; s2 += v * v;
        }
        s = wave_sum(s); s2 = wave_sum(s2);
        float mu  = s * (1.0f / D_);
        float var = s2 * (1.0f / D_) - mu * mu;
        float rs  = rsqrtf(var + LN_EPS_);
        for (int d = lane; d < D_; d += 32)
            xlns[r * D_ + d] =
                f2bf((xinf[r * D_ + d] - mu) * rs * lng[d] + lnb[d]);
    }
    __syncthreads();

    // ---- fragment lane geometry ----
    int arow  = lane & 15;               // A row (M)
    int khalf = (lane >> 4) << 3;        // A K-offset 0 / 8
    int col   = lane & 15;               // D col (N)
    int mhalf = (lane >> 4) << 3;        // D row offset 0 / 8
    int nt0   = wave * 3;

    // ================= GEMM1: h1 = silu(xln @ W1 + b1) =================
    {
        v8f acc[3] = {v8f{}, v8f{}, v8f{}};
        for (int kt = 0; kt < 12; ++kt) {
            Frag a;
#pragma unroll
            for (int v = 0; v < 8; ++v) {
                int kk = (khalf << 1) + (v << 1);
                a.u[v] = *(const unsigned int*)&xlns[arow * D_ + kt * 32 + kk];
            }
#pragma unroll
            for (int j = 0; j < 3; ++j) {
                Frag bf;
                bf.u = *(const v8u*)(w1p + ((size_t)((nt0 + j) * 12 + kt) * 32 + lane) * 8);
                acc[j] = __builtin_amdgcn_wmma_f32_16x16x32_bf16(
                    false, a.b, false, bf.b, (short)0, acc[j], false, false);
            }
        }
#pragma unroll
        for (int j = 0; j < 3; ++j) {
            int n = (nt0 + j) * 16 + col;
            float bias = kpb1[n];
#pragma unroll
            for (int r8 = 0; r8 < 8; ++r8) {
                int m = r8 + mhalf;
                h1s[m * D_ + n] = f2bf(silu(acc[j][r8] + bias));
            }
        }
    }
    __syncthreads();

    // ================= GEMM2 + epilogue ================================
    {
        v8f acc[3] = {v8f{}, v8f{}, v8f{}};
        for (int kt = 0; kt < 12; ++kt) {
            Frag a;
#pragma unroll
            for (int v = 0; v < 8; ++v) {
                int kk = (khalf << 1) + (v << 1);
                a.u[v] = *(const unsigned int*)&h1s[arow * D_ + kt * 32 + kk];
            }
#pragma unroll
            for (int j = 0; j < 3; ++j) {
                Frag bf;
                bf.u = *(const v8u*)(w2p + ((size_t)((nt0 + j) * 12 + kt) * 32 + lane) * 8);
                acc[j] = __builtin_amdgcn_wmma_f32_16x16x32_bf16(
                    false, a.b, false, bf.b, (short)0, acc[j], false, false);
            }
        }
#pragma unroll
        for (int j = 0; j < 3; ++j) {
            int n = (nt0 + j) * 16 + col;
            float b2 = kpb2[n];
            float ssum = 0.f;
#pragma unroll
            for (int r8 = 0; r8 < 8; ++r8) {
                int m = r8 + mhalf;
                float val = acc[j][r8] + b2
                          + xinf[m * D_ + n]
                          + lemb[row_y[m] * D_ + n];
                ssum += row_prob[m] * val;
            }
            ssum += __shfl_xor(ssum, 16, 32);   // combine the two row halves
            if (lane < 16)
                atomicAdd(&ctx_out[(size_t)b * D_ + n], ssum);
        }
    }
}

// ======================================================================
// Predictor: x = k + ctx ; h = silu(x @ pw1 + pb1) ; y = h @ pw2 + pb2
// block handles 16 rows of B.
// ======================================================================
__global__ void predictor_kernel(const float* __restrict__ k,
                                 const float* __restrict__ ctx,
                                 const unsigned int* __restrict__ pw1p,
                                 const float* __restrict__ pb1,
                                 const float* __restrict__ pw2,
                                 const float* __restrict__ pb2,
                                 float* __restrict__ out) {
    __shared__ __align__(16) unsigned short xb[16 * D_];   // bf16 x
    __shared__ __align__(16) float          hb[16 * D_];   // fp32 h
    int b0 = blockIdx.x * 16;
    int tid = threadIdx.x, wave = tid >> 5, lane = tid & 31;

    for (int i = tid; i < 16 * D_; i += 256) {
        int r = i / D_, d = i - r * D_;
        size_t g = (size_t)(b0 + r) * D_ + d;
        xb[i] = f2bf(k[g] + ctx[g]);
    }
    __syncthreads();

    int arow  = lane & 15;
    int khalf = (lane >> 4) << 3;
    int col   = lane & 15;
    int mhalf = (lane >> 4) << 3;
    int nt0   = wave * 3;

    {
        v8f acc[3] = {v8f{}, v8f{}, v8f{}};
        for (int kt = 0; kt < 12; ++kt) {
            Frag a;
#pragma unroll
            for (int v = 0; v < 8; ++v) {
                int kk = (khalf << 1) + (v << 1);
                a.u[v] = *(const unsigned int*)&xb[arow * D_ + kt * 32 + kk];
            }
#pragma unroll
            for (int j = 0; j < 3; ++j) {
                Frag bf;
                bf.u = *(const v8u*)(pw1p + ((size_t)((nt0 + j) * 12 + kt) * 32 + lane) * 8);
                acc[j] = __builtin_amdgcn_wmma_f32_16x16x32_bf16(
                    false, a.b, false, bf.b, (short)0, acc[j], false, false);
            }
        }
#pragma unroll
        for (int j = 0; j < 3; ++j) {
            int n = (nt0 + j) * 16 + col;
            float bias = pb1[n];
#pragma unroll
            for (int r8 = 0; r8 < 8; ++r8) {
                int m = r8 + mhalf;
                hb[m * D_ + n] = silu(acc[j][r8] + bias);
            }
        }
    }
    __syncthreads();

    // tiny head: 16 rows x 10 classes, wave-parallel dot over D
    for (int t = wave; t < 16 * NCLS_; t += 8) {
        int r = t / NCLS_, cls = t - r * NCLS_;
        float s = 0.f;
        for (int d = lane; d < D_; d += 32)
            s += hb[r * D_ + d] * pw2[d * NCLS_ + cls];
        s = wave_sum(s);
        if (lane == 0) out[(size_t)(b0 + r) * NCLS_ + cls] = s + pb2[cls];
    }
}

// ======================================================================
// launch
// ======================================================================
extern "C" void kernel_launch(void* const* d_in, const int* in_sizes, int n_in,
                              void* d_out, int out_size, void* d_ws, size_t ws_size,
                              hipStream_t stream) {
    (void)in_sizes; (void)n_in; (void)out_size; (void)ws_size;

    const float* k    = (const float*)d_in[0];
    const int*   cidx = (const int*)  d_in[1];
    const float* cand = (const float*)d_in[2];
    const int*   cy   = (const int*)  d_in[3];
    const float* lemb = (const float*)d_in[4];
    const float* lng  = (const float*)d_in[5];
    const float* lnb  = (const float*)d_in[6];
    const float* kpw1 = (const float*)d_in[7];
    const float* kpb1 = (const float*)d_in[8];
    const float* kpw2 = (const float*)d_in[9];
    const float* kpb2 = (const float*)d_in[10];
    const float* pw1  = (const float*)d_in[11];
    const float* pb1  = (const float*)d_in[12];
    const float* pw2  = (const float*)d_in[13];
    const float* pb2  = (const float*)d_in[14];
    float* out = (float*)d_out;

    // workspace layout (256B-aligned offsets)
    char* ws = (char*)d_ws;
    constexpr size_t OFF_PROBS = 0;                                    // B*C f32
    constexpr size_t OFF_CN    = OFF_PROBS + (size_t)B_ * C_ * 4;      // N f32
    constexpr size_t OFF_CTX   = OFF_CN    + (size_t)N_ * 4;           // B*D f32
    constexpr size_t OFF_W1P   = OFF_CTX   + (size_t)B_ * D_ * 4;      // packed bf16
    constexpr size_t PACKB     = (size_t)24 * 12 * 32 * 8 * 4;         // 294912 B
    constexpr size_t OFF_W2P   = OFF_W1P + PACKB;
    constexpr size_t OFF_PW1P  = OFF_W2P + PACKB;

    float*        probs = (float*)(ws + OFF_PROBS);
    float*        cn    = (float*)(ws + OFF_CN);
    float*        ctx   = (float*)(ws + OFF_CTX);
    unsigned int* w1p   = (unsigned int*)(ws + OFF_W1P);
    unsigned int* w2p   = (unsigned int*)(ws + OFF_W2P);
    unsigned int* pw1p  = (unsigned int*)(ws + OFF_PW1P);

    hipMemsetAsync(ctx, 0, (size_t)B_ * D_ * sizeof(float), stream);

    pack_w_kernel<<<288, 256, 0, stream>>>(kpw1, w1p);
    pack_w_kernel<<<288, 256, 0, stream>>>(kpw2, w2p);
    pack_w_kernel<<<288, 256, 0, stream>>>(pw1, pw1p);

    cand_norms_kernel<<<(N_ + 7) / 8, 256, 0, stream>>>(cand, cn);
    sim_softmax_kernel<<<B_, 256, 0, stream>>>(k, cidx, cand, cn, probs);

    values_kernel<<<B_ * 6, 256, 0, stream>>>(k, cidx, cand, cy, lemb,
                                              lng, lnb, kpb1, kpb2,
                                              w1p, w2p, probs, ctx);

    predictor_kernel<<<B_ / 16, 256, 0, stream>>>(k, ctx, pw1p, pb1, pw2, pb2, out);
}
